// PointNet2_79061757985223
// MI455X (gfx1250) — matrix operand order repaired and checked by reading, not compiled, not run on hardware
//
#include <hip/hip_runtime.h>
#include <hip/hip_bf16.h>

// ---------------------------------------------------------------------------
// PointNet++ forward for MI455X (gfx1250, wave32, WMMA).
// All linear layers run through v_wmma_f32_16x16x32_f16 with f16 operands
// staged in LDS (ds_load_b128 fragment fetches, register-cached A tiles)
// and f32 accumulation. FPS / ball-query / 3-NN are wave-cooperative VALU.
// ---------------------------------------------------------------------------

typedef __attribute__((ext_vector_type(16))) _Float16 v16h;
typedef __attribute__((ext_vector_type(8)))  float    v8f;
typedef __attribute__((ext_vector_type(4)))  unsigned v4u;
typedef __attribute__((__vector_size__(16))) int      v4i_vs;

#define DEVINL __device__ __forceinline__

#ifndef TRY_ASYNC_LDS
#define TRY_ASYNC_LDS 1
#endif
#if TRY_ASYNC_LDS && defined(__has_builtin)
#if __has_builtin(__builtin_amdgcn_global_load_async_to_lds_b128) && \
    __has_builtin(__builtin_amdgcn_s_wait_asynccnt)
#define HAVE_ASYNC_LDS 1
#endif
#endif

// ---------------- WMMA fragment builders (contiguous k order: k = 16*(lane/16) + s) ----------------

// A-matrix 16x32 f16: lane L<16 -> row L, kpairs {0..7};
// lane L>=16 -> row L-16, kpairs {8..15}. Each 4-kpair run is one
// contiguous 16B block -> 2x ds_load_b128.
DEVINL v16h frag_a(const _Float16* actRow, int ldK, int lane) {
  union { v16h h; v4u v[2]; } r;
  const int g = lane >> 4;
  const v4u* p = (const v4u*)(actRow + (size_t)(lane & 15) * ldK);
  r.v[0] = p[2 * g];      // kpairs g*8 .. g*8+3
  r.v[1] = p[2 * g + 1];  // kpairs g*8+4 .. g*8+7
  return r.h;
}

// B-matrix 32x16 f16 from fragment-major packed weights: each (kc,ct) tile is
// 32 lanes x 8 u32, so a lane's fragment is 32 contiguous bytes -> 2x b128.
DEVINL v16h frag_b(const unsigned* tile, int lane) {
  union { v16h h; v4u v[2]; } r;
  const v4u* p = (const v4u*)(tile + (size_t)lane * 8);
  r.v[0] = p[0];
  r.v[1] = p[1];
  return r.h;
}

DEVINL unsigned packh2(float a, float b) {
  union { _Float16 h[2]; unsigned u; } r;
  r.h[0] = (_Float16)a; r.h[1] = (_Float16)b;
  return r.u;
}

// Cooperative load of global f32 weights [kinReal x NOUT] into LDS as packed
// f16 pairs in FRAGMENT-MAJOR order:
//   wbuf[((kc*CT + ct)*32 + lane)*8 + v] = pack(W[2*kp][n], W[2*kp+1][n])
//   kp = kc*16 + v + (lane>>4)*8,  n = ct*16 + (lane&15)
template <int KINPAD, int NOUT>
DEVINL void load_weights(const float* __restrict__ W, int kinReal,
                         const float* __restrict__ b,
                         unsigned* wbuf, float* biasLds, int tid, int nthreads) {
  constexpr int CT = NOUT / 16;
  for (int i = tid; i < (KINPAD / 2) * NOUT; i += nthreads) {
    const int v = i & 7;
    const int lane = (i >> 3) & 31;
    const int tile = i >> 8;  // kc*CT + ct
    const int ct = tile % CT;
    const int kc = tile / CT;
    const int kp = kc * 16 + v + ((lane >> 4) << 3);
    const int n = ct * 16 + (lane & 15);
    const int k0 = 2 * kp, k1 = k0 + 1;
    float w0 = (k0 < kinReal) ? W[(size_t)k0 * NOUT + n] : 0.f;
    float w1 = (k1 < kinReal) ? W[(size_t)k1 * NOUT + n] : 0.f;
    wbuf[i] = packh2(w0, w1);
  }
  for (int i = tid; i < NOUT; i += nthreads) biasLds[i] = b[i];
}

// Core GEMM for one wave's 32 rows: register-cache all A fragments once,
// then per column tile load B once and feed both row-tile WMMAs.
template <int KIN, int NOUT, class Emit>
DEVINL void wmma_core(const _Float16* in, int ldIn, const unsigned* wbuf,
                      int waveRowBase, int lane, Emit&& emit) {
  constexpr int KC = KIN / 32, CT = NOUT / 16;
  v16h a[2][KC];
#pragma unroll
  for (int rt = 0; rt < 2; ++rt)
#pragma unroll
    for (int kc = 0; kc < KC; ++kc)
      a[rt][kc] = frag_a(in + (size_t)(waveRowBase + rt * 16) * ldIn + kc * 32,
                         ldIn, lane);
#pragma unroll
  for (int ct = 0; ct < CT; ++ct) {
    v8f acc0 = {0.f, 0.f, 0.f, 0.f, 0.f, 0.f, 0.f, 0.f};
    v8f acc1 = {0.f, 0.f, 0.f, 0.f, 0.f, 0.f, 0.f, 0.f};
#pragma unroll
    for (int kc = 0; kc < KC; ++kc) {
      v16h b = frag_b(wbuf + (size_t)(kc * CT + ct) * 256, lane);
      acc0 = __builtin_amdgcn_wmma_f32_16x16x32_f16(false, a[0][kc], false, b,
                                                    (short)0, acc0, false, false);
      acc1 = __builtin_amdgcn_wmma_f32_16x16x32_f16(false, a[1][kc], false, b,
                                                    (short)0, acc1, false, false);
    }
    emit(ct, acc0, acc1);
  }
}

// Linear + relu, f16 result into LDS.
template <int KIN, int NOUT>
DEVINL void wmma_layer(const _Float16* in, int ldIn, const unsigned* wbuf,
                       const float* bias, _Float16* out, int ldOut,
                       int waveRowBase, int lane) {
  wmma_core<KIN, NOUT>(in, ldIn, wbuf, waveRowBase, lane,
      [&](int ct, const v8f& acc0, const v8f& acc1) {
        const int col = ct * 16 + (lane & 15);
        const int g = lane >> 4;
        const float bc = bias[col];
#pragma unroll
        for (int r = 0; r < 8; ++r) {
          float v0 = acc0[r] + bc; v0 = v0 > 0.f ? v0 : 0.f;
          float v1 = acc1[r] + bc; v1 = v1 > 0.f ? v1 : 0.f;
          out[(size_t)(waveRowBase + r + g * 8) * ldOut + col] = (_Float16)v0;
          out[(size_t)(waveRowBase + 16 + r + g * 8) * ldOut + col] = (_Float16)v1;
        }
      });
}

// Linear + relu, f32 result straight to global (rows relative to block).
template <int KIN, int NOUT>
DEVINL void wmma_layer_global(const _Float16* in, int ldIn, const unsigned* wbuf,
                              const float* bias, float* __restrict__ outG,
                              int waveRowBase, int lane) {
  wmma_core<KIN, NOUT>(in, ldIn, wbuf, waveRowBase, lane,
      [&](int ct, const v8f& acc0, const v8f& acc1) {
        const int col = ct * 16 + (lane & 15);
        const int g = lane >> 4;
        const float bc = bias[col];
#pragma unroll
        for (int r = 0; r < 8; ++r) {
          float v0 = acc0[r] + bc; v0 = v0 > 0.f ? v0 : 0.f;
          float v1 = acc1[r] + bc; v1 = v1 > 0.f ? v1 : 0.f;
          outG[(size_t)(waveRowBase + r + g * 8) * NOUT + col] = v0;
          outG[(size_t)(waveRowBase + 16 + r + g * 8) * NOUT + col] = v1;
        }
      });
}

// Final SA layer: linear + relu then max over the wave's 32 rows (neighbors).
template <int KIN, int NOUT>
DEVINL void wmma_layer_maxpool(const _Float16* in, int ldIn, const unsigned* wbuf,
                               const float* bias, float* __restrict__ outRow,
                               int waveRowBase, int lane) {
  wmma_core<KIN, NOUT>(in, ldIn, wbuf, waveRowBase, lane,
      [&](int ct, const v8f& acc0, const v8f& acc1) {
        const float bc = bias[ct * 16 + (lane & 15)];
        float m = -3.0e38f;
#pragma unroll
        for (int r = 0; r < 8; ++r) {
          float v0 = acc0[r] + bc; v0 = v0 > 0.f ? v0 : 0.f;
          float v1 = acc1[r] + bc; v1 = v1 > 0.f ? v1 : 0.f;
          m = fmaxf(m, fmaxf(v0, v1));
        }
        m = fmaxf(m, __shfl_xor(m, 16, 32));  // merge wave32 lane halves
        if (lane < 16) outRow[ct * 16 + lane] = m;
      });
}

// ------------------------------ FPS ----------------------------------------
template <int DIM>
__global__ __launch_bounds__(256) void fps_kernel(const float* __restrict__ pts,
                                                  int N, int npoint,
                                                  int* __restrict__ idxOut) {
  const int b = blockIdx.x;
  const int tid = threadIdx.x, lane = tid & 31, w = tid >> 5;
  extern __shared__ char smem[];
  float* mind = (float*)smem;              // [N]
  float* lastPt = mind + N;                // [DIM]
  float* redV = lastPt + DIM;              // [8]
  int* redI = (int*)(redV + 8);            // [8]
  int* sLast = redI + 8;                   // [1]

  const float* P = pts + (size_t)b * N * DIM;
  for (int i = tid; i < N; i += 256) mind[i] = 1.0e10f;
  if (tid == 0) *sLast = 0;
  __syncthreads();

  for (int it = 0; it < npoint; ++it) {
    const int last = *sLast;
    if (tid == 0) idxOut[(size_t)b * npoint + it] = last;
    for (int k = tid; k < DIM; k += 256) lastPt[k] = P[(size_t)last * DIM + k];
    __syncthreads();

    float bv = -1.f; int bi = 0;
    for (int i = tid; i < N; i += 256) {
      float d = 0.f;
#pragma unroll 4
      for (int k = 0; k < DIM; ++k) {
        float t = P[(size_t)i * DIM + k] - lastPt[k];
        d += t * t;
      }
      float m = mind[i];
      m = m < d ? m : d;
      mind[i] = m;
      if (m > bv) { bv = m; bi = i; }
    }
#pragma unroll
    for (int off = 16; off > 0; off >>= 1) {
      float ov = __shfl_xor(bv, off, 32);
      int oi = __shfl_xor(bi, off, 32);
      if (ov > bv || (ov == bv && oi < bi)) { bv = ov; bi = oi; }
    }
    if (lane == 0) { redV[w] = bv; redI[w] = bi; }
    __syncthreads();
    if (tid == 0) {
      float v = redV[0]; int ii = redI[0];
      for (int q = 1; q < 8; ++q)
        if (redV[q] > v || (redV[q] == v && redI[q] < ii)) { v = redV[q]; ii = redI[q]; }
      *sLast = ii;
    }
    __syncthreads();
  }
}

// --------------------- Set abstraction (fused) -------------------------------
template <int DIM, int K0, int H1, int H2, int H3, int CPB>
__global__ __launch_bounds__(CPB * 32) void sa_kernel(
    const float* __restrict__ pts, int N, const int* __restrict__ idx, int npoint,
    float radius2,
    const float* __restrict__ W1, const float* __restrict__ b1,
    const float* __restrict__ W2, const float* __restrict__ b2,
    const float* __restrict__ W3, const float* __restrict__ b3,
    float* __restrict__ out) {
  constexpr int ROWS = CPB * 32;
  constexpr int LD = (K0 > H1 ? (K0 > H2 ? K0 : H2) : (H1 > H2 ? H1 : H2));
  constexpr int WA = (K0 / 2) * H1, WB = (H1 / 2) * H2, WC = (H2 / 2) * H3;
  constexpr int WMAX = (WA > WB ? (WA > WC ? WA : WC) : (WB > WC ? WB : WC));
  constexpr int HMAX = (H1 > H2 ? (H1 > H3 ? H1 : H3) : (H2 > H3 ? H2 : H3));

  const int b = blockIdx.y;
  const int centerBase = blockIdx.x * CPB;
  const int tid = threadIdx.x, lane = tid & 31, w = tid >> 5;
  const int nthreads = CPB * 32;

  extern __shared__ char smem[];
  _Float16* bufA = (_Float16*)smem;                           // [ROWS][LD]
  _Float16* bufB = bufA + (size_t)ROWS * LD;                  // [ROWS][LD]
  unsigned* wbuf = (unsigned*)(bufB + (size_t)ROWS * LD);     // [WMAX]
  float* biasLds = (float*)(wbuf + WMAX);                     // [HMAX]
  int* gidx = (int*)(biasLds + HMAX);                         // [CPB][32]
  float* centerF = (float*)(gidx + CPB * 32);                 // [CPB][DIM]

  const float* P = pts + (size_t)b * N * DIM;

  // center features -> LDS
  {
    const int ci = idx[(size_t)b * npoint + centerBase + w];
    for (int k = lane; k < DIM; k += 32)
      centerF[w * DIM + k] = P[(size_t)ci * DIM + k];
  }
  __syncthreads();

  // ball query: first 32 in-radius points in index order, pad with first hit
  {
    const float* C = centerF + w * DIM;
    int count = 0, first = -1;
    for (int base = 0; base < N && count < 32; base += 32) {
      const int p = base + lane;
      __builtin_prefetch(&P[(size_t)(base + 32) * DIM]);  // global_prefetch_b8
      float d = 0.f;
#pragma unroll 4
      for (int k = 0; k < DIM; ++k) {
        float t = P[(size_t)p * DIM + k] - C[k];
        d += t * t;
      }
      const bool hit = (p < N) && (d <= radius2);
      const unsigned mask = (unsigned)__ballot(hit);
      if (first < 0 && mask) first = base + (__ffs((int)mask) - 1);
      const int pos = count + __popc(mask & ((1u << lane) - 1u));
      if (hit && pos < 32) gidx[w * 32 + pos] = p;
      count += __popc(mask);
    }
    if (count > 32) count = 32;
    if (first < 0) first = 0;
    if (lane >= count) gidx[w * 32 + lane] = first;
  }
  __syncthreads();

  // gather + center-subtract into bufA (f16, ld = LD, zero-pad to K0)
  {
    const int row = w * 32 + lane;
    const int p = gidx[w * 32 + lane];
    const float* C = centerF + w * DIM;
#pragma unroll 4
    for (int k = 0; k < DIM; ++k)
      bufA[(size_t)row * LD + k] = (_Float16)(P[(size_t)p * DIM + k] - C[k]);
    for (int k = DIM; k < K0; ++k) bufA[(size_t)row * LD + k] = (_Float16)0.f;
  }

  __syncthreads();
  load_weights<K0, H1>(W1, DIM, b1, wbuf, biasLds, tid, nthreads);
  __syncthreads();
  wmma_layer<K0, H1>(bufA, LD, wbuf, biasLds, bufB, LD, w * 32, lane);
  __syncthreads();
  load_weights<H1, H2>(W2, H1, b2, wbuf, biasLds, tid, nthreads);
  __syncthreads();
  wmma_layer<H1, H2>(bufB, LD, wbuf, biasLds, bufA, LD, w * 32, lane);
  __syncthreads();
  load_weights<H2, H3>(W3, H2, b3, wbuf, biasLds, tid, nthreads);
  __syncthreads();
  wmma_layer_maxpool<H2, H3>(bufA, LD, wbuf, biasLds,
                             out + ((size_t)b * npoint + centerBase + w) * H3,
                             w * 32, lane);
}

// --------------------- Feature propagation: 3-NN search ---------------------
template <int DIM>
__global__ __launch_bounds__(256) void fp_nn_kernel(
    const float* __restrict__ posDense, int Nd,
    const float* __restrict__ src, int srcN,
    const int* __restrict__ sIdx, int Ns,
    int* __restrict__ nnOut, float* __restrict__ wOut) {
  const int b = blockIdx.y;
  const int i = blockIdx.x * 256 + threadIdx.x;
  if (i >= Nd) return;
  const float* PD = posDense + ((size_t)b * Nd + i) * DIM;
  const float* SB = src + (size_t)b * srcN * DIM;
  const int* SI = sIdx + (size_t)b * Ns;

  float d0 = 3e38f, d1 = 3e38f, d2 = 3e38f;
  int i0 = 0, i1 = 0, i2 = 0;
  for (int s = 0; s < Ns; ++s) {
    const float* PS = SB + (size_t)SI[s] * DIM;
    float d = 0.f;
#pragma unroll 4
    for (int k = 0; k < DIM; ++k) {
      float t = PD[k] - PS[k];
      d += t * t;
    }
    if (d < d0)      { d2 = d1; i2 = i1; d1 = d0; i1 = i0; d0 = d; i0 = s; }
    else if (d < d1) { d2 = d1; i2 = i1; d1 = d;  i1 = s; }
    else if (d < d2) { d2 = d;  i2 = s; }
  }
  float w0 = 1.f / fmaxf(d0, 1e-10f);
  float w1 = 1.f / fmaxf(d1, 1e-10f);
  float w2 = 1.f / fmaxf(d2, 1e-10f);
  const float inv = 1.f / (w0 + w1 + w2);
  const size_t o = ((size_t)b * Nd + i) * 3;
  nnOut[o + 0] = i0; nnOut[o + 1] = i1; nnOut[o + 2] = i2;
  wOut[o + 0] = w0 * inv; wOut[o + 1] = w1 * inv; wOut[o + 2] = w2 * inv;
}

// ---------------- Feature propagation: interp + linear + relu ----------------
template <int CIN, int COUT>
__global__ __launch_bounds__(128) void fp_interp_kernel(
    const float* __restrict__ feats, int Ns,
    const int* __restrict__ nn, const float* __restrict__ wgt, int Nd,
    const float* __restrict__ W, const float* __restrict__ bias,
    float* __restrict__ out) {
  const int b = blockIdx.y;
  const int rowBase = blockIdx.x * 128;
  const int tid = threadIdx.x, lane = tid & 31, w = tid >> 5;

  extern __shared__ char smem[];
  _Float16* X = (_Float16*)smem;                              // [128][CIN]
  unsigned* wbuf = (unsigned*)(X + (size_t)128 * CIN);        // [(CIN/2)*COUT]
  float* biasLds = (float*)(wbuf + (CIN / 2) * COUT);         // [COUT]

  {  // build interpolated row tid
    const size_t gi = (size_t)b * Nd + rowBase + tid;
    const int* np = nn + gi * 3;
    const float* wp = wgt + gi * 3;
    const float* F = feats + (size_t)b * Ns * CIN;
    const float* f0 = F + (size_t)np[0] * CIN;
    const float* f1 = F + (size_t)np[1] * CIN;
    const float* f2 = F + (size_t)np[2] * CIN;
    const float w0 = wp[0], w1 = wp[1], w2 = wp[2];
#pragma unroll 4
    for (int k = 0; k < CIN; ++k)
      X[(size_t)tid * CIN + k] = (_Float16)(w0 * f0[k] + w1 * f1[k] + w2 * f2[k]);
  }
  load_weights<CIN, COUT>(W, CIN, bias, wbuf, biasLds, tid, 128);
  __syncthreads();
  wmma_layer_global<CIN, COUT>(X, CIN, wbuf, biasLds,
                               out + ((size_t)b * Nd + rowBase) * COUT,
                               w * 32, lane);
}

// ------------------------------ FC head --------------------------------------
__global__ __launch_bounds__(128) void fc_kernel(
    const float* __restrict__ l4, int Nd,
    const float* __restrict__ W1, const float* __restrict__ b1,  // 64x128
    const float* __restrict__ W2, const float* __restrict__ b2,  // 128x2
    float* __restrict__ out) {
  const int b = blockIdx.y;
  const int rowBase = blockIdx.x * 128;
  const int tid = threadIdx.x, lane = tid & 31, w = tid >> 5;

  extern __shared__ char smem[];
  _Float16* X = (_Float16*)smem;              // [128][64]
  _Float16* H = X + 128 * 64;                 // [128][128]
  unsigned* wbuf = (unsigned*)(H + 128 * 128);// [32*128]
  float* biasLds = (float*)(wbuf + 32 * 128); // [128]
  float* w2l = biasLds + 128;                 // [256]
  float* b2l = w2l + 256;                     // [2]

  // Stage fc2 weights (raw f32) into LDS — async DMA path when available.
#if defined(HAVE_ASYNC_LDS)
  if (tid < 64) {
    __builtin_amdgcn_global_load_async_to_lds_b128(
        (__attribute__((address_space(1))) v4i_vs*)(W2 + tid * 4),
        (__attribute__((address_space(3))) v4i_vs*)(w2l + tid * 4),
        0, 0);
  }
  __builtin_amdgcn_s_wait_asynccnt(0);
#else
  for (int i = tid; i < 256; i += 128) w2l[i] = W2[i];
#endif
  if (tid < 2) b2l[tid] = b2[tid];

  {
    const float* srcRow = l4 + ((size_t)b * Nd + rowBase + tid) * 64;
#pragma unroll 4
    for (int k = 0; k < 64; ++k) X[(size_t)tid * 64 + k] = (_Float16)srcRow[k];
  }
  load_weights<64, 128>(W1, 64, b1, wbuf, biasLds, tid, 128);
  __syncthreads();
  wmma_layer<64, 128>(X, 64, wbuf, biasLds, H, 128, w * 32, lane);
  __syncthreads();
  {
    float s0 = b2l[0], s1 = b2l[1];
#pragma unroll 4
    for (int k = 0; k < 128; ++k) {
      const float h = (float)H[(size_t)tid * 128 + k];
      s0 += h * w2l[k * 2 + 0];
      s1 += h * w2l[k * 2 + 1];
    }
    float* o = out + ((size_t)b * Nd + rowBase + tid) * 2;
    o[0] = s0; o[1] = s1;
  }
}

// ------------------------------ host side ------------------------------------

extern "C" void kernel_launch(void* const* d_in, const int* in_sizes, int n_in,
                              void* d_out, int out_size, void* d_ws, size_t ws_size,
                              hipStream_t stream) {
  (void)in_sizes; (void)n_in; (void)out_size; (void)ws_size;
  constexpr int B = 4, N = 16384, NP1 = 1024, NP2 = 256;

  // Input order: xyz, then params flattened in dict insertion order
  // (sa1 W/b x3, sa2 W/b x3, fp1 W/b, fp2 W/b, fc1 W/b, fc2 W/b).
  const float* xyz = (const float*)d_in[0];
  const float* s1W1 = (const float*)d_in[1];  const float* s1b1 = (const float*)d_in[2];
  const float* s1W2 = (const float*)d_in[3];  const float* s1b2 = (const float*)d_in[4];
  const float* s1W3 = (const float*)d_in[5];  const float* s1b3 = (const float*)d_in[6];
  const float* s2W1 = (const float*)d_in[7];  const float* s2b1 = (const float*)d_in[8];
  const float* s2W2 = (const float*)d_in[9];  const float* s2b2 = (const float*)d_in[10];
  const float* s2W3 = (const float*)d_in[11]; const float* s2b3 = (const float*)d_in[12];
  const float* fp1W = (const float*)d_in[13]; const float* fp1b = (const float*)d_in[14];
  const float* fp2W = (const float*)d_in[15]; const float* fp2b = (const float*)d_in[16];
  const float* fc1W = (const float*)d_in[17]; const float* fc1b = (const float*)d_in[18];
  const float* fc2W = (const float*)d_in[19]; const float* fc2b = (const float*)d_in[20];
  float* outp = (float*)d_out;

  // workspace layout (256B aligned)
  char* ws = (char*)d_ws;
  size_t off = 0;
  auto take = [&](size_t bytes) { char* p = ws + off; off = (off + bytes + 255) & ~(size_t)255; return p; };
  int*   idx1 = (int*)  take((size_t)B * NP1 * 4);
  int*   idx2 = (int*)  take((size_t)B * NP2 * 4);
  float* l1   = (float*)take((size_t)B * NP1 * 128 * 4);
  float* l2   = (float*)take((size_t)B * NP2 * 256 * 4);
  float* l3   = (float*)take((size_t)B * NP1 * 128 * 4);
  float* l4   = (float*)take((size_t)B * N * 64 * 4);
  int*   nn1  = (int*)  take((size_t)B * NP1 * 3 * 4);
  float* wg1  = (float*)take((size_t)B * NP1 * 3 * 4);
  int*   nn2  = (int*)  take((size_t)B * N * 3 * 4);
  float* wg2  = (float*)take((size_t)B * N * 3 * 4);

  // --- SA1 ---
  {
    const size_t lds = (size_t)N * 4 + 3 * 4 + 8 * 4 + 8 * 4 + 4 + 128;
    fps_kernel<3><<<B, 256, lds, stream>>>(xyz, N, NP1, idx1);
  }
  {
    const size_t lds = (size_t)256 * 64 * 2 * 2 + 4096 * 4 + 128 * 4 +
                       8 * 32 * 4 + 8 * 3 * 4 + 128;
    sa_kernel<3, 32, 64, 64, 128, 8><<<dim3(NP1 / 8, B), 256, lds, stream>>>(
        xyz, N, idx1, NP1, 0.2f * 0.2f, s1W1, s1b1, s1W2, s1b2, s1W3, s1b3, l1);
  }
  // --- SA2 ---
  {
    const size_t lds = (size_t)NP1 * 4 + 128 * 4 + 8 * 4 + 8 * 4 + 4 + 128;
    fps_kernel<128><<<B, 256, lds, stream>>>(l1, NP1, NP2, idx2);
  }
  {
    const size_t lds = (size_t)128 * 128 * 2 * 2 + 16384 * 4 + 256 * 4 +
                       4 * 32 * 4 + 4 * 128 * 4 + 128;
    sa_kernel<128, 128, 128, 128, 256, 4><<<dim3(NP2 / 4, B), 128, lds, stream>>>(
        l1, NP1, idx2, NP2, 0.4f * 0.4f, s2W1, s2b1, s2W2, s2b2, s2W3, s2b3, l2);
  }
  // --- FP1: interp l2 over l1 positions (128-dim), 256->128 ---
  fp_nn_kernel<128><<<dim3(NP1 / 256, B), 256, 0, stream>>>(
      l1, NP1, l1, NP1, idx2, NP2, nn1, wg1);
  {
    const size_t lds = (size_t)128 * 256 * 2 + (256 / 2) * 128 * 4 + 128 * 4 + 128;
    fp_interp_kernel<256, 128><<<dim3(NP1 / 128, B), 128, lds, stream>>>(
        l2, NP2, nn1, wg1, NP1, fp1W, fp1b, l3);
  }
  // --- FP2: interp l3 over xyz positions (3-dim), 128->64 ---
  fp_nn_kernel<3><<<dim3(N / 256, B), 256, 0, stream>>>(
      xyz, N, xyz, N, idx1, NP1, nn2, wg2);
  {
    const size_t lds = (size_t)128 * 128 * 2 + (128 / 2) * 64 * 4 + 64 * 4 + 128;
    fp_interp_kernel<128, 64><<<dim3(N / 128, B), 128, lds, stream>>>(
        l3, NP1, nn2, wg2, N, fp2W, fp2b, l4);
  }
  // --- FC head: 64 -> 128 (relu) -> 2 ---
  {
    const size_t lds = (size_t)128 * 64 * 2 + (size_t)128 * 128 * 2 +
                       32 * 128 * 4 + 128 * 4 + 256 * 4 + 2 * 4 + 128;
    fc_kernel<<<dim3(N / 128, B), 128, lds, stream>>>(
        l4, N, fc1W, fc1b, fc2W, fc2b, outp);
  }
}
